// MultiSinglePeriodCrossAttention_63823214018612
// MI455X (gfx1250) — matrix operand order repaired and checked by reading, not compiled, not run on hardware
//
#include <hip/hip_runtime.h>
#include <hip/hip_bf16.h>

typedef __attribute__((ext_vector_type(16))) _Float16 v16h;
typedef __attribute__((ext_vector_type(4)))  _Float16 h4;
typedef __attribute__((ext_vector_type(8)))  float    v8f;

#define NB   8      // batch
#define CCH  128    // channels C
#define DM   128    // d_model D
#define NHH  8      // heads
#define HDD  16     // head dim
#define NSEQ 16384  // H*W
#define EPSF 1e-6f

#define NTP  128    // proj/final n-tile (LDS = 32KB A + 32KB B)
#define NIP  2      // proj: tiles per workgroup (weights staged once per 256 cols)
#define NTA  64     // attn n-tile
#define NIA  4      // attn: tiles per workgroup
#define NKV  64     // kv-reduce sub-tile
#define NIK  2      // kv-reduce: sub-tiles per workgroup

__device__ __forceinline__ int lane_id() { return threadIdx.x & 31; }
__device__ __forceinline__ int wave_id() { return threadIdx.x >> 5; }

// ---- WMMA fragment-layout position helpers (offset within a 512-half block) ----
// A (16x32, 16-bit): lanes 0-15: M=lane, K=0..15; lanes 16-31: M=lane-16, K=16..31
__device__ __forceinline__ int a_pos(int m, int k) {
  int lane = (m & 15) + ((k & 16) ? 16 : 0);
  int e    = k & 15;
  return lane * 16 + e;
}
// B (32x16, 16-bit): lanes 0-15: N=lane, K=0..15; lanes 16-31: N=lane-16, K=16..31
__device__ __forceinline__ int b_pos(int k, int n) {
  int lane = (n & 15) + ((k & 16) ? 16 : 0);
  int e    = k & 15;
  return lane * 16 + e;
}
// fragment load: 16 contiguous halves per lane (2x ds_load_b128)
__device__ __forceinline__ v16h load_frag(const _Float16* block) {
  return *(const v16h*)(block + lane_id() * 16);
}

// ---- stage a [128 x 128] weight matrix (row stride wstride) into A-frag layout ----
template <typename T>
__device__ __forceinline__ void stage_w_afrag(const T* __restrict__ W, int wstride,
                                              _Float16* __restrict__ sA, int tid) {
  for (int q = tid; q < DM * DM / 4; q += 256) {
    int r  = q >> 5;            // 32 quads per row
    int c0 = (q & 31) * 4;
    const T* src = W + (size_t)r * wstride + c0;
    h4 v = { (_Float16)src[0], (_Float16)src[1], (_Float16)src[2], (_Float16)src[3] };
    _Float16* dst = sA + ((r >> 4) * 4 + (c0 >> 5)) * 512 + a_pos(r & 15, c0 & 31);
    *(h4*)dst = v;
  }
}

// ---- stage a [128 x NTC] activation tile (row stride NSEQ) into B-frag layout ----
// contraction k = channel row, column n; dst blocks: (kt*(NTC/16) + nt) * 512
template <typename T, int NTC>
__device__ __forceinline__ void stage_x_bfrag(const T* __restrict__ X,
                                              _Float16* __restrict__ sB, int tid) {
  const int n = tid & (NTC - 1);
  for (int kq = (tid / NTC); kq < 32; kq += (256 / NTC)) {
    int k0 = kq * 4;
    h4 v;
#pragma unroll
    for (int j = 0; j < 4; ++j) v[j] = (_Float16)X[(size_t)(k0 + j) * NSEQ + n];
    _Float16* dst = sB + ((k0 >> 5) * (NTC / 16) + (n >> 4)) * 512 + b_pos(k0 & 31, n);
    *(h4*)dst = v;
  }
}

// ---------------- zero init for reduction buffers ----------------
__global__ void zero_kernel(float* __restrict__ p, int n) {
  int i = blockIdx.x * blockDim.x + threadIdx.x;
  if (i < n) p[i] = 0.0f;
}

// ---------------- projection GEMM: Out[b,d,n] = act(sum_c W[d,c] X[b,c,n]) ----------------
__global__ __launch_bounds__(256)
void proj_gemm_kernel(const float* __restrict__ X, const float* __restrict__ W,
                      _Float16* __restrict__ Out, int apply_relu) {
  __shared__ __align__(32) _Float16 sA[8 * 4 * 512];           // 32 KB, A-frag W
  __shared__ __align__(32) _Float16 sB[4 * (NTP / 16) * 512];  // 32 KB, B-frag X
  const int strips = NSEQ / (NTP * NIP);
  const int b  = blockIdx.x / strips;
  const int n0 = (blockIdx.x % strips) * (NTP * NIP);
  const int tid = threadIdx.x;
  const int w  = wave_id();
  const int l  = lane_id();
  const int n  = l & 15;
  const int mb = (l < 16) ? 0 : 8;

  stage_w_afrag(W, CCH, sA, tid);
  __syncthreads();
  v16h afr[4];
#pragma unroll
  for (int kt = 0; kt < 4; ++kt) afr[kt] = load_frag(sA + (w * 4 + kt) * 512);

  for (int it = 0; it < NIP; ++it) {
    const float* Xb = X + (size_t)b * CCH * NSEQ + n0 + it * NTP;
    __syncthreads();                       // previous tile's B reads done
    stage_x_bfrag<float, NTP>(Xb, sB, tid);
    __builtin_prefetch(Xb + NTP, 0, 1);    // global_prefetch_b8
    __syncthreads();

    _Float16* OutB = Out + (size_t)b * DM * NSEQ + n0 + it * NTP;
#pragma unroll
    for (int nt = 0; nt < NTP / 16; ++nt) {
      v8f acc = {};
#pragma unroll
      for (int kt = 0; kt < 4; ++kt) {
        v16h bfr = load_frag(sB + (kt * (NTP / 16) + nt) * 512);
        acc = __builtin_amdgcn_wmma_f32_16x16x32_f16(false, afr[kt], false, bfr,
                                                     (short)0, acc, false, false);
      }
#pragma unroll
      for (int r = 0; r < 8; ++r) {
        float v = acc[r];
        if (apply_relu) v = fmaxf(v, 0.0f) + EPSF;
        OutB[(size_t)(w * 16 + mb + r) * NSEQ + nt * 16 + n] = (_Float16)v;
      }
    }
  }
}

// ---------------- kv & ksum reduction per head ----------------
// kvmat[b][h][d][m] += sum_n KH[b][16h+d][n] * VH[b][16h+m][n];  ksum[b][h][d] += sum_n KH
__global__ __launch_bounds__(256)
void kv_reduce_kernel(const _Float16* __restrict__ KH, const _Float16* __restrict__ VH,
                      float* __restrict__ kvmat, float* __restrict__ ksum) {
  __shared__ __align__(32) _Float16 sKa[NHH * 2 * 512];  // 16 KB, A-frag K per head
  __shared__ __align__(32) _Float16 sVb[NHH * 2 * 512];  // 16 KB, B-frag V^T per head
  __shared__ float sKS[DM];
  const int strips = NSEQ / (NKV * NIK);
  const int b  = blockIdx.x / strips;
  const int n0 = (blockIdx.x % strips) * (NKV * NIK);
  const int tid = threadIdx.x;
  const int h = wave_id();  // 8 waves <-> 8 heads

  if (tid < DM) sKS[tid] = 0.0f;
  v8f acc = {};

  for (int it = 0; it < NIK; ++it) {
    const _Float16* Kb = KH + (size_t)b * DM * NSEQ + n0 + it * NKV;
    const _Float16* Vb = VH + (size_t)b * DM * NSEQ + n0 + it * NKV;
    __syncthreads();
    // stage K (A-frag: i=d, k=n) with fused ksum partial via LDS float atomics
    for (int q = tid; q < DM * NKV / 4; q += 256) {
      int r  = q >> 4;            // 16 quads per 64-col row
      int c0 = (q & 15) * 4;
      h4 v = *(const h4*)(Kb + (size_t)r * NSEQ + c0);
      atomicAdd(&sKS[r], (float)v[0] + (float)v[1] + (float)v[2] + (float)v[3]);
      _Float16* dst = sKa + ((r >> 4) * 2 + (c0 >> 5)) * 512 + a_pos(r & 15, c0 & 31);
      *(h4*)dst = v;
    }
    // stage V transposed into B-frag: k = n (contraction), col = m
    for (int q = tid; q < DM * NKV / 4; q += 256) {
      int r  = q >> 4;
      int c0 = (q & 15) * 4;
      h4 v = *(const h4*)(Vb + (size_t)r * NSEQ + c0);
      _Float16* dst = sVb + ((r >> 4) * 2 + (c0 >> 5)) * 512 + b_pos(c0 & 31, r & 15);
      *(h4*)dst = v;
    }
    __syncthreads();
#pragma unroll
    for (int kt = 0; kt < NKV / 32; ++kt) {
      v16h afr = load_frag(sKa + (h * 2 + kt) * 512);
      v16h bfr = load_frag(sVb + (h * 2 + kt) * 512);
      acc = __builtin_amdgcn_wmma_f32_16x16x32_f16(false, afr, false, bfr,
                                                   (short)0, acc, false, false);
    }
  }

  // single atomic flush per workgroup
  float* kvb = kvmat + ((size_t)b * NHH + h) * 256;
  const int l  = lane_id();
  const int n  = l & 15;
  const int mb = (l < 16) ? 0 : 8;
#pragma unroll
  for (int r = 0; r < 8; ++r) atomicAdd(&kvb[(mb + r) * 16 + n], acc[r]);
  __syncthreads();
  if (tid < DM) atomicAdd(&ksum[(size_t)b * DM + tid], sKS[tid]);
}

// ---------------- fold kv into Wback: Weff[b][c][16h+d] = sum_m Wback[c][16h+m]*kv[b][h][d][m] ----------------
__global__ __launch_bounds__(256)
void weff_kernel(const float* __restrict__ Wback, const float* __restrict__ kvmat,
                 _Float16* __restrict__ Weff) {
  __shared__ float skv[NHH * 256];
  const int b = blockIdx.x;
  const int tid = threadIdx.x;
  for (int i = tid; i < NHH * 256; i += 256) skv[i] = kvmat[(size_t)b * NHH * 256 + i];
  __syncthreads();
  for (int o = tid; o < DM * DM; o += 256) {
    int c = o >> 7, col = o & 127;
    int h = col >> 4, d = col & 15;
    const float* wb = Wback + c * DM + h * 16;
    const float* kv = skv + h * 256 + d * 16;
    float s = 0.0f;
#pragma unroll
    for (int m = 0; m < 16; ++m) s += wb[m] * kv[m];
    Weff[((size_t)b * DM + c) * DM + col] = (_Float16)s;
  }
}

// ---------------- one cross-attention call: out = base + Weff @ (QH * z) ----------------
__global__ __launch_bounds__(256)
void attn_y_kernel(const _Float16* __restrict__ QH, const float* __restrict__ ksum,
                   const _Float16* __restrict__ Weff, const float* __restrict__ base,
                   float* __restrict__ out) {
  __shared__ __align__(32) _Float16 sA[8 * 4 * 512];           // 32 KB, A-frag Weff
  __shared__ __align__(32) _Float16 sQ[4 * (NTA / 16) * 512];  // 16 KB, B-frag qh
  __shared__ float sKS[NHH * HDD];
  const int strips = NSEQ / (NTA * NIA);
  const int b  = blockIdx.x / strips;
  const int n0 = (blockIdx.x % strips) * (NTA * NIA);
  const int tid = threadIdx.x;
  const int w  = wave_id();
  const int l  = lane_id();
  const int n  = l & 15;
  const int mb = (l < 16) ? 0 : 8;

  stage_w_afrag(Weff + (size_t)b * DM * DM, DM, sA, tid);
  if (tid < NHH * HDD) sKS[tid] = ksum[(size_t)b * NHH * HDD + tid];
  __syncthreads();
  v16h afr[4];
#pragma unroll
  for (int kt = 0; kt < 4; ++kt) afr[kt] = load_frag(sA + (w * 4 + kt) * 512);

  for (int it = 0; it < NIA; ++it) {
    const _Float16* Qb = QH + (size_t)b * DM * NSEQ + n0 + it * NTA;
    __syncthreads();
    stage_x_bfrag<_Float16, NTA>(Qb, sQ, tid);
    __syncthreads();

    // z-normalize in place: in B-frag layout the 16 head-dims of one column are
    // contiguous in one lane => one v16h LDS read-modify-write per (head, column).
    for (int i = tid; i < NHH * NTA; i += 256) {
      int h = i / NTA, col = i & (NTA - 1);
      _Float16* p = sQ + (((h >> 1) * (NTA / 16) + (col >> 4)) * 512 +
                          ((col & 15) + ((h & 1) ? 16 : 0)) * 16);
      v16h q = *(v16h*)p;
      float s = 0.0f;
#pragma unroll
      for (int d = 0; d < HDD; ++d) s += (float)q[d] * sKS[h * HDD + d];
      float z = 1.0f / (s + EPSF);
#pragma unroll
      for (int d = 0; d < HDD; ++d) q[d] = (_Float16)((float)q[d] * z);
      *(v16h*)p = q;
    }
    __syncthreads();

    const size_t gofs = (size_t)b * CCH * NSEQ + n0 + it * NTA;
    const float* baseB = base + gofs;
    float* outB = out + gofs;
#pragma unroll
    for (int nt = 0; nt < NTA / 16; ++nt) {
      v8f acc = {};
#pragma unroll
      for (int kt = 0; kt < 4; ++kt) {
        v16h bfr = load_frag(sQ + (kt * (NTA / 16) + nt) * 512);
        acc = __builtin_amdgcn_wmma_f32_16x16x32_f16(false, afr[kt], false, bfr,
                                                     (short)0, acc, false, false);
      }
#pragma unroll
      for (int r = 0; r < 8; ++r) {
        size_t off = (size_t)(w * 16 + mb + r) * NSEQ + nt * 16 + n;
        outB[off] = baseB[off] + acc[r];
      }
    }
  }
}

// ---------------- final: out = Wout @ concat(T0,T1,T2) + Xf  (K = 3*128 in 3 chunks) ----------------
__global__ __launch_bounds__(256)
void final_kernel(const float* __restrict__ T0, const float* __restrict__ T1,
                  const float* __restrict__ T2, const float* __restrict__ Wout,
                  const float* __restrict__ Xf, float* __restrict__ outp) {
  __shared__ __align__(32) _Float16 sA[8 * 4 * 512];           // 32 KB
  __shared__ __align__(32) _Float16 sB[4 * (NTP / 16) * 512];  // 32 KB
  const int strips = NSEQ / NTP;
  const int b  = blockIdx.x / strips;
  const int n0 = (blockIdx.x % strips) * NTP;
  const int tid = threadIdx.x;
  const int w  = wave_id();

  v8f acc[NTP / 16] = {};
  const float* Ts[3] = {T0, T1, T2};
  for (int chunk = 0; chunk < 3; ++chunk) {
    __syncthreads();
    stage_w_afrag(Wout + chunk * CCH, 3 * CCH, sA, tid);
    const float* Tb = Ts[chunk] + (size_t)b * CCH * NSEQ + n0;
    stage_x_bfrag<float, NTP>(Tb, sB, tid);
    __builtin_prefetch(Tb + NTP, 0, 1);
    __syncthreads();
    v16h afr[4];
#pragma unroll
    for (int kt = 0; kt < 4; ++kt) afr[kt] = load_frag(sA + (w * 4 + kt) * 512);
#pragma unroll
    for (int nt = 0; nt < NTP / 16; ++nt) {
#pragma unroll
      for (int kt = 0; kt < 4; ++kt) {
        v16h bfr = load_frag(sB + (kt * (NTP / 16) + nt) * 512);
        acc[nt] = __builtin_amdgcn_wmma_f32_16x16x32_f16(false, afr[kt], false, bfr,
                                                         (short)0, acc[nt], false, false);
      }
    }
  }
  const size_t gofs = (size_t)b * CCH * NSEQ + n0;
  const float* XfB = Xf + gofs;
  float* oB = outp + gofs;
  const int l  = lane_id();
  const int n  = l & 15;
  const int mb = (l < 16) ? 0 : 8;
#pragma unroll
  for (int nt = 0; nt < NTP / 16; ++nt)
#pragma unroll
    for (int r = 0; r < 8; ++r) {
      size_t off = (size_t)(w * 16 + mb + r) * NSEQ + nt * 16 + n;
      oB[off] = XfB[off] + acc[nt][r];
    }
}

// ---------------- host orchestration ----------------
extern "C" void kernel_launch(void* const* d_in, const int* in_sizes, int n_in,
                              void* d_out, int out_size, void* d_ws, size_t ws_size,
                              hipStream_t stream) {
  (void)in_sizes; (void)n_in; (void)out_size; (void)ws_size;
  const float* t1    = (const float*)d_in[0];
  const float* t2    = (const float*)d_in[1];
  const float* t3    = (const float*)d_in[2];
  const float* fs    = (const float*)d_in[3];
  const float* Wq    = (const float*)d_in[4];
  const float* Wk    = (const float*)d_in[5];
  const float* Wv    = (const float*)d_in[6];
  const float* Wback = (const float*)d_in[7];
  const float* Wout  = (const float*)d_in[8];
  float* outp = (float*)d_out;

  char* ws = (char*)d_ws;
  size_t off = 0;
  auto alloc = [&](size_t bytes) -> void* {
    void* p = ws + off;
    off += (bytes + 255) & ~(size_t)255;
    return p;
  };
  const size_t seq = (size_t)NB * DM * NSEQ;  // 16,777,216 elements
  _Float16* QH0 = (_Float16*)alloc(seq * 2);
  _Float16* QH1 = (_Float16*)alloc(seq * 2);
  _Float16* QH2 = (_Float16*)alloc(seq * 2);
  _Float16* KH  = (_Float16*)alloc(seq * 2);
  _Float16* VH  = (_Float16*)alloc(seq * 2);
  float* TA0 = (float*)alloc(seq * 4);
  float* TA1 = (float*)alloc(seq * 4);
  float* TA2 = (float*)alloc(seq * 4);
  float* kv      = (float*)alloc((size_t)4 * NB * NHH * 256 * 4);
  float* ksum    = (float*)alloc((size_t)4 * NB * NHH * HDD * 4);
  _Float16* Weff = (_Float16*)alloc((size_t)4 * NB * DM * DM * 2);

  const int g_proj = NB * (NSEQ / (NTP * NIP));  // 512
  const int g_kv   = NB * (NSEQ / (NKV * NIK));  // 1024
  const int g_attn = NB * (NSEQ / (NTA * NIA));  // 512
  const int g_fin  = NB * (NSEQ / NTP);          // 1024
  dim3 blk(256);

  const int zeroN = 4 * NB * NHH * 256 + 4 * NB * NHH * HDD;
  zero_kernel<<<(zeroN + 255) / 256, blk, 0, stream>>>(kv, zeroN);

  // kv-side reductions for t1, t2, t3, fusion
  const float* Xs[4] = {t1, t2, t3, fs};
  for (int j = 0; j < 4; ++j) {
    proj_gemm_kernel<<<g_proj, blk, 0, stream>>>(Xs[j], Wk, KH, 1);
    proj_gemm_kernel<<<g_proj, blk, 0, stream>>>(Xs[j], Wv, VH, 0);
    kv_reduce_kernel<<<g_kv, blk, 0, stream>>>(KH, VH,
        kv + (size_t)j * NB * NHH * 256, ksum + (size_t)j * NB * NHH * HDD);
  }
  for (int j = 0; j < 4; ++j)
    weff_kernel<<<NB, blk, 0, stream>>>(Wback, kv + (size_t)j * NB * NHH * 256,
                                        Weff + (size_t)j * NB * DM * DM);

  // stage-A query projections (shared across both calls per query)
  proj_gemm_kernel<<<g_proj, blk, 0, stream>>>(t1, Wq, QH0, 1);
  proj_gemm_kernel<<<g_proj, blk, 0, stream>>>(t2, Wq, QH1, 1);
  proj_gemm_kernel<<<g_proj, blk, 0, stream>>>(t3, Wq, QH2, 1);

  auto ksj = [&](int j) { return ksum + (size_t)j * NB * NHH * HDD; };
  auto wfj = [&](int j) { return Weff + (size_t)j * NB * DM * DM; };

  // stage A: t_ia = t_i + Y(i,j) + Y(i,k)
  attn_y_kernel<<<g_attn, blk, 0, stream>>>(QH0, ksj(1), wfj(1), t1,  TA0);
  attn_y_kernel<<<g_attn, blk, 0, stream>>>(QH0, ksj(2), wfj(2), TA0, TA0);
  attn_y_kernel<<<g_attn, blk, 0, stream>>>(QH1, ksj(0), wfj(0), t2,  TA1);
  attn_y_kernel<<<g_attn, blk, 0, stream>>>(QH1, ksj(2), wfj(2), TA1, TA1);
  attn_y_kernel<<<g_attn, blk, 0, stream>>>(QH2, ksj(0), wfj(0), t3,  TA2);
  attn_y_kernel<<<g_attn, blk, 0, stream>>>(QH2, ksj(1), wfj(1), TA2, TA2);

  // stage B: re-project queries from t_ia, attend to fusion (j=3), in-place residual
  proj_gemm_kernel<<<g_proj, blk, 0, stream>>>(TA0, Wq, QH0, 1);
  proj_gemm_kernel<<<g_proj, blk, 0, stream>>>(TA1, Wq, QH1, 1);
  proj_gemm_kernel<<<g_proj, blk, 0, stream>>>(TA2, Wq, QH2, 1);
  attn_y_kernel<<<g_attn, blk, 0, stream>>>(QH0, ksj(3), wfj(3), TA0, TA0);
  attn_y_kernel<<<g_attn, blk, 0, stream>>>(QH1, ksj(3), wfj(3), TA1, TA1);
  attn_y_kernel<<<g_attn, blk, 0, stream>>>(QH2, ksj(3), wfj(3), TA2, TA2);

  // final: Wout @ concat + fs, straight into [B,C,H,W]
  final_kernel<<<g_fin, blk, 0, stream>>>(TA0, TA1, TA2, Wout, fs, outp);
}
